// graph_constructor_661424963839
// MI455X (gfx1250) — matrix operand order repaired and checked by reading, not compiled, not run on hardware
//
#include <hip/hip_runtime.h>
#include <hip/hip_bf16.h>

// ---------------------------------------------------------------------------
// Problem constants (from reference): N=8192 rows, D=64 features, K=32 top-k
// ---------------------------------------------------------------------------
#define NROWS 8192
#define DIM   64
#define TOPK  32
#define ALPHA 3.0f

typedef __attribute__((ext_vector_type(16))) _Float16 v16h;
typedef __attribute__((ext_vector_type(8)))  _Float16 v8h;
typedef __attribute__((ext_vector_type(8)))  float    v8f;

// Hardware tanh (CDNA5 TRANS32 op). ISA: one independent op / V_NOP required
// after a trans op before its result is consumed -> embed v_nop.
__device__ __forceinline__ float htanh(float x) {
    float r;
    asm("v_tanh_f32 %0, %1\n\tv_nop" : "=v"(r) : "v"(x));
    return r;
}

// ---------------------------------------------------------------------------
// Kernel 1: nv = tanh(ALPHA * (emb[idx] @ W^T + b)), stored as f16 for WMMA.
// One 64-thread block per row; emb row broadcast through LDS.
// ---------------------------------------------------------------------------
__global__ __launch_bounds__(64) void nv_kernel(const int* __restrict__ idx,
                                                const float* __restrict__ emb,
                                                const float* __restrict__ W,
                                                const float* __restrict__ bias,
                                                _Float16* __restrict__ out) {
    const int row = blockIdx.x;
    const int t   = threadIdx.x;           // output feature 0..63
    __shared__ float e[DIM];
    e[t] = emb[(size_t)idx[row] * DIM + t];
    __syncthreads();
    float acc = bias[t];
    const float* wrow = W + (size_t)t * DIM;
#pragma unroll
    for (int k = 0; k < DIM; ++k) acc += e[k] * wrow[k];
    out[(size_t)row * DIM + t] = (_Float16)htanh(ALPHA * acc);
}

// ---------------------------------------------------------------------------
// Kernel 2: adj = relu(tanh(ALPHA * (nv1@nv2^T - nv2@nv1^T)))
// 256-thread block = 8 waves; block computes a 128x128 tile; each wave owns a
// 16x128 row strip (8 WMMA tiles of 16x16, K=64 as 2 steps of 32).
// v_wmma_f32_16x16x32_f16; nv operands (2 MB) stay L2-resident.
// ---------------------------------------------------------------------------
__global__ __launch_bounds__(256) void adj_gemm(const _Float16* __restrict__ nv1,
                                                const _Float16* __restrict__ nv2,
                                                float* __restrict__ out) {
    const int lane = threadIdx.x & 31;
    const int wave = threadIdx.x >> 5;            // 0..7
    const int i0   = blockIdx.y * 128 + wave * 16; // row base of this wave
    const int j0   = blockIdx.x * 128;             // col base of this block

    const int mn = lane & 15;        // row (A) / col (B,C) within 16-tile
    const int hi = lane >> 4;        // lane-group select per ISA VGPR layouts

    v8f acc1[8], acc2[8];
    const v8f vzero = {};
#pragma unroll
    for (int t = 0; t < 8; ++t) { acc1[t] = vzero; acc2[t] = vzero; }

    // A operand: contiguous k order (not the ISA table order) — lane m holds
    // halves[0..15] = K hi*16+0..15 (matches B)
    const _Float16* a1row = nv1 + (size_t)(i0 + mn) * DIM;
    const _Float16* a2row = nv2 + (size_t)(i0 + mn) * DIM;

#pragma unroll
    for (int kk = 0; kk < 2; ++kk) {
        const int k0 = kk * 32;
        union { v16h v; v8h h[2]; } ua1, ua2;
        ua1.v = *(const v16h*)(a1row + k0 + hi * 16);
        ua2.v = *(const v16h*)(a2row + k0 + hi * 16);
        const v16h a1 = ua1.v;
        const v16h a2 = ua2.v;

#pragma unroll
        for (int t = 0; t < 8; ++t) {
            const int col = j0 + t * 16 + mn;
            // B, same contiguous k order as A: lane n holds halves = B[K hi*16+0..15][n]
            //   B[k][n] = nvX[col][k]  -> 32 contiguous bytes of one nv row.
            const v16h b2 = *(const v16h*)(nv2 + (size_t)col * DIM + k0 + hi * 16);
            const v16h b1 = *(const v16h*)(nv1 + (size_t)col * DIM + k0 + hi * 16);
            acc1[t] = __builtin_amdgcn_wmma_f32_16x16x32_f16(
                false, a1, false, b2, (short)0, acc1[t], false, false);
            acc2[t] = __builtin_amdgcn_wmma_f32_16x16x32_f16(
                false, a2, false, b1, (short)0, acc2[t], false, false);
        }
    }

    // Epilogue: a = acc1 - acc2 ; adj = relu(tanh(ALPHA*a)) via v_tanh_f32.
    // C/D layout: VGPR e -> M = e + (hi?8:0), N = mn.
#pragma unroll
    for (int t = 0; t < 8; ++t) {
        const int col = j0 + t * 16 + mn;
#pragma unroll
        for (int e = 0; e < 8; ++e) {
            const int m = e + hi * 8;
            float a = acc1[t][e] - acc2[t][e];
            float r = fmaxf(htanh(ALPHA * a), 0.0f);
            out[(size_t)(i0 + m) * NROWS + col] = r;
        }
    }
}

// ---------------------------------------------------------------------------
// Kernel 3: row-wise top-K mask. One 256-thread block per row. Row cached in
// LDS; two-level 1024-bin LDS histogram gives a 2^-20-resolution threshold
// for the K-th largest value in [0,1] (adj = relu(tanh(.)) is in [0,1]).
// In-place on d_out (regenerated every call by adj_gemm -> deterministic).
// ---------------------------------------------------------------------------
__global__ __launch_bounds__(256) void topk_kernel(float* __restrict__ adj) {
    const int row = blockIdx.x;
    const int tid = threadIdx.x;

    __shared__ float    s_row[NROWS];
    __shared__ unsigned s_hist[1024];
    __shared__ int      s_bstar;
    __shared__ unsigned s_need;
    __shared__ float    s_thresh;

    float* g = adj + (size_t)row * NROWS;

    // Load row into LDS (float4 vectorized) + zero histogram.
    const float4* src = (const float4*)g;
    float4* rowv = (float4*)s_row;
    for (int i = tid; i < NROWS / 4; i += 256) rowv[i] = src[i];
    for (int i = tid; i < 1024; i += 256) s_hist[i] = 0u;
    __syncthreads();

    // Level-1 histogram: bin = floor(v*1024), clamped (tanh can round to 1.0f).
    for (int i = tid; i < NROWS; i += 256) {
        float v = s_row[i];
        int b = (int)(v * 1024.0f);
        b = (b > 1023) ? 1023 : b;
        atomicAdd(&s_hist[b], 1u);
    }
    __syncthreads();

    if (tid == 0) {
        unsigned cum = 0;
        int b = 1023;
        for (; b > 0; --b) {
            unsigned c = s_hist[b];
            if (cum + c >= (unsigned)TOPK) break;
            cum += c;
        }
        s_bstar = b;
        s_need  = (unsigned)TOPK - cum;   // slots to fill from bin b
    }
    __syncthreads();
    const int bstar = s_bstar;
    const unsigned need = s_need;
    __syncthreads();

    // Level-2: sub-histogram inside bin bstar (reuse s_hist).
    for (int i = tid; i < 1024; i += 256) s_hist[i] = 0u;
    __syncthreads();
    const float fb = (float)bstar;
    for (int i = tid; i < NROWS; i += 256) {
        float v = s_row[i];
        int b = (int)(v * 1024.0f);
        b = (b > 1023) ? 1023 : b;
        if (b == bstar) {
            int s = (int)((v * 1024.0f - fb) * 1024.0f);
            s = (s < 0) ? 0 : ((s > 1023) ? 1023 : s);
            atomicAdd(&s_hist[s], 1u);
        }
    }
    __syncthreads();

    if (tid == 0) {
        unsigned cum = 0;
        int s = 1023;
        for (; s > 0; --s) {
            unsigned c = s_hist[s];
            if (cum + c >= need) break;
            cum += c;
        }
        s_thresh = (fb + (float)s * (1.0f / 1024.0f)) * (1.0f / 1024.0f);
    }
    __syncthreads();
    const float T = s_thresh;

    // Write masked row back (float4 vectorized).
    float4* dst = (float4*)g;
    for (int i = tid; i < NROWS / 4; i += 256) {
        float4 v = rowv[i];
        v.x = (v.x >= T) ? v.x : 0.0f;
        v.y = (v.y >= T) ? v.y : 0.0f;
        v.z = (v.z >= T) ? v.z : 0.0f;
        v.w = (v.w >= T) ? v.w : 0.0f;
        dst[i] = v;
    }
}

// ---------------------------------------------------------------------------
// Launch: inputs in setup_inputs() order: idx, emb1, emb2, W1, b1, W2, b2
// ---------------------------------------------------------------------------
extern "C" void kernel_launch(void* const* d_in, const int* in_sizes, int n_in,
                              void* d_out, int out_size, void* d_ws, size_t ws_size,
                              hipStream_t stream) {
    const int*   idx  = (const int*)d_in[0];
    const float* emb1 = (const float*)d_in[1];
    const float* emb2 = (const float*)d_in[2];
    const float* W1   = (const float*)d_in[3];
    const float* b1   = (const float*)d_in[4];
    const float* W2   = (const float*)d_in[5];
    const float* b2   = (const float*)d_in[6];
    float* out = (float*)d_out;

    _Float16* nv1h = (_Float16*)d_ws;                       // 1 MB
    _Float16* nv2h = nv1h + (size_t)NROWS * DIM;            // 1 MB

    nv_kernel<<<NROWS, 64, 0, stream>>>(idx, emb1, W1, b1, nv1h);
    nv_kernel<<<NROWS, 64, 0, stream>>>(idx, emb2, W2, b2, nv2h);

    dim3 grid(NROWS / 128, NROWS / 128);                    // 64 x 64 tiles
    adj_gemm<<<grid, 256, 0, stream>>>(nv1h, nv2h, out);

    topk_kernel<<<NROWS, 256, 0, stream>>>(out);
}
